// LinearAttention_28656021799604
// MI455X (gfx1250) — matrix operand order repaired and checked by reading, not compiled, not run on hardware
//
#include <hip/hip_runtime.h>

#define B_ 4
#define H_ 16
#define T_ 2048
#define D_ 64
#define QTILE 128  // 8 waves x 16-row strips
#define NWAVES 8
#define KCH 32
#define KSTR 66    // K tile row stride in shorts (33-bank stride: conflict-free)
#define VSTR 34    // V tile (transposed) row stride in shorts (17-bank stride)
#define SSTR4 34   // per-wave S tile row stride in floats (even -> 8B aligned pairs)

typedef __attribute__((ext_vector_type(16))) __bf16   v16bf;
typedef __attribute__((ext_vector_type(8)))  float    v8f;
typedef __attribute__((ext_vector_type(8)))  unsigned v8u;

// Pack two f32 into two bf16 (truncation) with a single v_perm_b32.
static __device__ __forceinline__ unsigned pack_bf2(float lo, float hi) {
  return __builtin_amdgcn_perm(__builtin_bit_cast(unsigned, hi),
                               __builtin_bit_cast(unsigned, lo), 0x07060302u);
}
static __device__ __forceinline__ unsigned short bf_hi(float f) {
  return (unsigned short)(__builtin_bit_cast(unsigned, f) >> 16);  // ds_store_b16_d16_hi
}

#define WMMA_BF16(A, Bm, C) \
  __builtin_amdgcn_wmma_f32_16x16x32_bf16(false, __builtin_bit_cast(v16bf, A), \
                                          false, __builtin_bit_cast(v16bf, Bm), \
                                          (short)0, C, false, false)

__global__ __launch_bounds__(256, 2)
void lin_attn_causal(const float* __restrict__ Q,
                     const float* __restrict__ K,
                     const float* __restrict__ V,
                     float* __restrict__ O) {
  __shared__ __align__(16) unsigned short Kt[2][KCH * KSTR];        // [key][d] bf16, dbl-buf
  __shared__ __align__(16) unsigned short Vt[2][D_ * VSTR];         // [d][key] bf16, dbl-buf
  __shared__ __align__(16) float          Stf[NWAVES][16 * SSTR4];  // per-wave S [row][key] f32

  const int tid  = threadIdx.x;
  const int lane = tid & 31;
  const int wave = tid >> 5;            // 0..7
  const int ln   = lane & 15;
  const int hf   = lane >> 4;

  const int qbase = blockIdx.x * QTILE;
  const int bh    = blockIdx.y;
  const int rowb  = qbase + wave * 16;  // this wave's 16-row strip

  const size_t headoff = (size_t)bh * T_ * D_;
  const float* Qh = Q + headoff;
  const float* Kh = K + headoff;
  const float* Vh = V + headoff;
  float*       Oh = O + headoff;

  // ---- Q A-fragments (two 16x32 bf16) straight from global, perm-packed ----
  v8u aq0u, aq1u;
  {
    const float* qrow = Qh + (size_t)(rowb + ln) * D_;
#pragma unroll
    for (int p = 0; p < 8; ++p) {
      const int k = 2 * p + 16 * hf;  // A-layout k index
      float2 v0 = *(const float2*)(qrow + k);
      float2 v1 = *(const float2*)(qrow + 32 + k);
      aq0u[p] = pack_bf2(v0.x, v0.y);
      aq1u[p] = pack_bf2(v1.x, v1.y);
    }
  }

  v8f o0 = {}, o1 = {}, o2 = {}, o3 = {};

  // cooperative staging split: 256 threads, 8 threads/key, 8 d-values each
  const int key  = tid >> 3;          // 0..31
  const int dst0 = (tid & 7) * 8;     // d start (two float4)
  const int nchunks = (qbase + QTILE) / KCH;   // causal: only chunks <= diagonal

  // ---- prologue: stage chunk 0 into buffer 0 ----
  {
    const float* ksrc = Kh + (size_t)key * D_ + dst0;
    const float* vsrc = Vh + (size_t)key * D_ + dst0;
#pragma unroll
    for (int i = 0; i < 2; ++i) {
      float4 kf = *(const float4*)(ksrc + 4 * i);
      float4 vf = *(const float4*)(vsrc + 4 * i);
      const int d = dst0 + 4 * i;
      *(unsigned*)&Kt[0][key * KSTR + d]     = pack_bf2(kf.x, kf.y);
      *(unsigned*)&Kt[0][key * KSTR + d + 2] = pack_bf2(kf.z, kf.w);
      Vt[0][(d+0) * VSTR + key] = bf_hi(vf.x);
      Vt[0][(d+1) * VSTR + key] = bf_hi(vf.y);
      Vt[0][(d+2) * VSTR + key] = bf_hi(vf.z);
      Vt[0][(d+3) * VSTR + key] = bf_hi(vf.w);
    }
  }

  for (int j = 0; j < nchunks; ++j) {
    const int kb  = j * KCH;
    const int buf = j & 1;

    __syncthreads();   // chunk j staged & previous reads of buf^1 complete

    // ---- issue chunk j+1 global loads NOW (latency hidden under compute) ----
    const bool more = (j + 1 < nchunks);
    float4 kf0, kf1, vf0, vf1;
    if (more) {
      const float* ksrc = Kh + (size_t)(kb + KCH + key) * D_ + dst0;
      const float* vsrc = Vh + (size_t)(kb + KCH + key) * D_ + dst0;
      kf0 = *(const float4*)(ksrc + 0);  vf0 = *(const float4*)(vsrc + 0);
      kf1 = *(const float4*)(ksrc + 4);  vf1 = *(const float4*)(vsrc + 4);
      if (j + 2 < nchunks) {            // global_prefetch_b8 two chunks ahead
        __builtin_prefetch(ksrc + (size_t)KCH * D_, 0, 1);
        __builtin_prefetch(vsrc + (size_t)KCH * D_, 0, 1);
      }
    }

    // ---- compute chunk j (wave-uniform causal skip) ----
    if (kb <= rowb + 15) {
      const unsigned short* Ktb = Kt[buf];
      const unsigned short* Vtb = Vt[buf];

      // load ALL K B-fragments first, then back-to-back WMMAs
      v8u bk00, bk01, bk10, bk11;
#pragma unroll
      for (int p = 0; p < 8; ++p) {
        const int e = 2 * p;                         // B-layout: k = e + 16*hf
        bk00[p] = *(const unsigned*)&Ktb[(0  + ln) * KSTR + 16 * hf + e];
        bk01[p] = *(const unsigned*)&Ktb[(0  + ln) * KSTR + 32 + 16 * hf + e];
        bk10[p] = *(const unsigned*)&Ktb[(16 + ln) * KSTR + 16 * hf + e];
        bk11[p] = *(const unsigned*)&Ktb[(16 + ln) * KSTR + 32 + 16 * hf + e];
      }
      v8f c0 = {}, c1 = {};
      c0 = WMMA_BF16(aq0u, bk00, c0);
      c1 = WMMA_BF16(aq0u, bk10, c1);
      c0 = WMMA_BF16(aq1u, bk01, c0);
      c1 = WMMA_BF16(aq1u, bk11, c1);

      // causal mask only on chunks straddling this wave's strip (wave-uniform)
      if (kb + 31 > rowb) {
        const int col0 = kb + ln, col1 = kb + 16 + ln;
#pragma unroll
        for (int r = 0; r < 8; ++r) {
          const int row = rowb + r + 8 * hf;
          c0[r] = (col0 <= row) ? c0[r] : 0.0f;
          c1[r] = (col1 <= row) ? c1[r] : 0.0f;
        }
      }
#pragma unroll
      for (int r = 0; r < 8; ++r) {
        Stf[wave][(r + 8 * hf) * SSTR4 + ln]      = c0[r];
        Stf[wave][(r + 8 * hf) * SSTR4 + 16 + ln] = c1[r];
      }

      // reload S as bf16 A fragment + load ALL V B-fragments, then 4 WMMAs
      v8u sau, vb0, vb1, vb2, vb3;
#pragma unroll
      for (int p = 0; p < 8; ++p) {
        const int k = 2 * p + 16 * hf;
        float2 sv = *(const float2*)&Stf[wave][ln * SSTR4 + k];
        sau[p] = pack_bf2(sv.x, sv.y);
        vb0[p] = *(const unsigned*)&Vtb[(0  + ln) * VSTR + 16 * hf + 2 * p];
        vb1[p] = *(const unsigned*)&Vtb[(16 + ln) * VSTR + 16 * hf + 2 * p];
        vb2[p] = *(const unsigned*)&Vtb[(32 + ln) * VSTR + 16 * hf + 2 * p];
        vb3[p] = *(const unsigned*)&Vtb[(48 + ln) * VSTR + 16 * hf + 2 * p];
      }
      o0 = WMMA_BF16(sau, vb0, o0);
      o1 = WMMA_BF16(sau, vb1, o1);
      o2 = WMMA_BF16(sau, vb2, o2);
      o3 = WMMA_BF16(sau, vb3, o3);
    }

    // ---- write staged chunk j+1 into the other buffer ----
    if (more) {
      const int nb = buf ^ 1;
#pragma unroll
      for (int i = 0; i < 2; ++i) {
        float4 kf = (i == 0) ? kf0 : kf1;
        float4 vf = (i == 0) ? vf0 : vf1;
        const int d = dst0 + 4 * i;
        *(unsigned*)&Kt[nb][key * KSTR + d]     = pack_bf2(kf.x, kf.y);
        *(unsigned*)&Kt[nb][key * KSTR + d + 2] = pack_bf2(kf.z, kf.w);
        Vt[nb][(d+0) * VSTR + key] = bf_hi(vf.x);
        Vt[nb][(d+1) * VSTR + key] = bf_hi(vf.y);
        Vt[nb][(d+2) * VSTR + key] = bf_hi(vf.z);
        Vt[nb][(d+3) * VSTR + key] = bf_hi(vf.w);
      }
    }
  }

  // ---- epilogue: C/D layout -> global fp32 ----
#pragma unroll
  for (int r = 0; r < 8; ++r) {
    float* orow = Oh + (size_t)(rowb + r + 8 * hf) * D_ + ln;
    orow[0]  = o0[r];
    orow[16] = o1[r];
    orow[32] = o2[r];
    orow[48] = o3[r];
  }
}

extern "C" void kernel_launch(void* const* d_in, const int* in_sizes, int n_in,
                              void* d_out, int out_size, void* d_ws, size_t ws_size,
                              hipStream_t stream) {
  (void)in_sizes; (void)n_in; (void)out_size; (void)d_ws; (void)ws_size;
  const float* q = (const float*)d_in[0];
  const float* k = (const float*)d_in[1];
  const float* v = (const float*)d_in[2];
  float* o = (float*)d_out;
  dim3 grid(T_ / QTILE, B_ * H_);   // 16 query tiles x 64 (b,h) heads = 1024 WGs
  lin_attn_causal<<<grid, dim3(256, 1, 1), 0, stream>>>(q, k, v, o);
}
